// SchNetAngular_2774548873992
// MI455X (gfx1250) — matrix operand order repaired and checked by reading, not compiled, not run on hardware
//
#include <hip/hip_runtime.h>
#include <hip/hip_bf16.h>

// ---------------------------------------------------------------------------
// SchNet-with-angular-features, fused for MI455X (gfx1250, wave32, WMMA).
// All GEMMs use v_wmma_f32_16x16x32_bf16 (bf16 in, fp32 accumulate).
// Filter network (10 GFLOP/layer dominant part) fused per-atom: the
// [B,A,N,F] filter tensor never touches HBM.
// B-matrices are stored transposed ([N][K], K contiguous) so WMMA B-fragments
// are two contiguous ds_load_b128s; all staging is uint4 (16B) vector copies.
// ---------------------------------------------------------------------------

typedef __bf16 bf16_t;
typedef __attribute__((ext_vector_type(16))) __bf16 v16bf;
typedef __attribute__((ext_vector_type(8)))  __bf16 v8bf;
typedef __attribute__((ext_vector_type(8)))  float  v8f;

#define Bn 8
#define An 512
#define Nn 64
#define Dd 128
#define Ff 128
#define GAn 512
#define BA (Bn*An)          // 4096
#define GPAD 32             // gaussians padded 25 -> 32 (K of GEMM1)

__device__ __forceinline__ float b2f(bf16_t h) { return (float)h; }
__device__ __forceinline__ bf16_t f2b(float f) { return (bf16_t)f; }

// shifted softplus: ln(1+e^x) - ln2, numerically stable
__device__ __forceinline__ float ssp(float x) {
    return fmaxf(x, 0.f) + __logf(1.f + __expf(-fabsf(x))) - 0.6931471805599453f;
}

// ---- WMMA fragment loads from LDS tiles (ISA 7.12.2 layouts, wave32) ------
// A (16x32 bf16, row-major tile, ld elements per row):
// lane l holds row M = l&15, half = l>>4; values are two contiguous runs:
//   a[0..7]  = row[8*half .. 8*half+7]          (K 0..7 / 8..15)
//   a[8..15] = row[16+8*half .. 16+8*half+7]    (K 16..23 / 24..31)
__device__ __forceinline__ v16bf load_afrag(const bf16_t* A, int ld, int mbase, int lane) {
    const int m  = mbase + (lane & 15);
    const int hh = lane >> 4;
    const bf16_t* p = A + (size_t)m * ld + 8 * hh;
    union { v16bf v; v8bf h[2]; } u;
    u.h[0] = *(const v8bf*)(p);
    u.h[1] = *(const v8bf*)(p + 16);
    return u.v;
}
// B (32x16 bf16) from a TRANSPOSED tile BtT[128][32] (n-major, K contiguous):
//   b[0..7] = BtT[n][8*half .. 8*half+7], b[8..15] = BtT[n][16+8*half .. 16+8*half+7]
__device__ __forceinline__ v16bf load_bfrag_t(const bf16_t* BtT, int nbase, int lane) {
    const int n  = nbase + (lane & 15);
    const int hh = lane >> 4;
    const bf16_t* p = BtT + (size_t)n * GPAD + 8 * hh;
    union { v16bf v; v8bf h[2]; } u;
    u.h[0] = *(const v8bf*)(p);
    u.h[1] = *(const v8bf*)(p + 16);
    return u.v;
}

__device__ __forceinline__ v8f wmma_bf16(v16bf a, v16bf b, v8f c) {
    return __builtin_amdgcn_wmma_f32_16x16x32_bf16(false, a, false, b, (short)0, c,
                                                   false, false);
}

// ---------------------------------------------------------------------------
// Prep kernels
// ---------------------------------------------------------------------------
__global__ __launch_bounds__(256) void embed_kernel(const int* __restrict__ z,
                                                    const float* __restrict__ emb,
                                                    float* __restrict__ xf,
                                                    bf16_t* __restrict__ xb) {
    const int idx = blockIdx.x * 256 + threadIdx.x;      // BA*D total
    const int ba = idx >> 7, d = idx & 127;
    const float v = emb[(size_t)z[ba] * Dd + d];
    xf[idx] = v;
    xb[idx] = f2b(v);
}

__global__ __launch_bounds__(256) void dist_gauss_kernel(const float* __restrict__ pos,
                                                         const int* __restrict__ nbr,
                                                         const int* __restrict__ msk,
                                                         bf16_t* __restrict__ fij,
                                                         float* __restrict__ scal) {
    const int pid = blockIdx.x * 256 + threadIdx.x;      // BA*N total
    const int ba = pid >> 6;
    const int b  = ba >> 9;
    const int j  = nbr[pid];
    const float* pi = pos + (size_t)ba * 3;
    const float* pj = pos + (size_t)(b * An + j) * 3;
    const float dx = pj[0] - pi[0], dy = pj[1] - pi[1], dz = pj[2] - pi[2];
    const float r = sqrtf(dx * dx + dy * dy + dz * dz + 1e-12f);
    const float width = (5.0f - 1.2f) / 24.0f;
    const float coef  = -0.5f / (width * width);
#pragma unroll
    for (int g = 0; g < GPAD; ++g) {
        float f = 0.f;
        if (g < 25) { const float d = r - (1.2f + width * (float)g); f = __expf(coef * d * d); }
        fij[(size_t)pid * GPAD + g] = f2b(f);
    }
    scal[pid] = ((r <= 5.0f) ? 1.f : 0.f) * (float)msk[pid];
}

// plain f32 -> bf16 (A-side matrices, kept row-major)
__global__ __launch_bounds__(256) void cvt_kernel(const float* __restrict__ s,
                                                  bf16_t* __restrict__ d, int n) {
    const int i = blockIdx.x * 256 + threadIdx.x;
    if (i < n) d[i] = f2b(s[i]);
}

// [nmat][K][N] f32 -> [nmat][N][K] bf16 (transposed for B-fragment loads)
__global__ __launch_bounds__(256) void cvt_t_kernel(const float* __restrict__ s,
                                                    bf16_t* __restrict__ d,
                                                    int K, int N, int total) {
    const int i = blockIdx.x * 256 + threadIdx.x;
    if (i >= total) return;
    const int mat = i / (K * N);
    const int rem = i - mat * (K * N);
    const int k = rem / N, n = rem - k * N;
    d[(size_t)mat * K * N + (size_t)n * K + k] = f2b(s[i]);
}

// filt_w1 [L,25,128] -> transposed padded bf16 [L,128,32]
__global__ __launch_bounds__(256) void cvt_pad_w1t_kernel(const float* __restrict__ s,
                                                          bf16_t* __restrict__ d) {
    const int i = blockIdx.x * 256 + threadIdx.x;        // 3*128*32 total
    const int l = i / (Ff * GPAD), r = i % (Ff * GPAD);
    const int n = r >> 5, g = r & 31;
    d[i] = f2b((g < 25) ? s[((size_t)l * 25 + g) * Ff + n] : 0.f);
}

// ---------------------------------------------------------------------------
// Staging helpers (all uint4 = 8 bf16 vector copies)
// ---------------------------------------------------------------------------
// A tile: 64x32 bf16 from row-major A[ldK], rows m0.., cols kc*32..
__device__ __forceinline__ void stage_A(bf16_t (*At)[32], const bf16_t* A, int ldK,
                                        int m0, int kc, int tid) {
    const int r = tid >> 2, q = tid & 3;                 // 256 threads, 1 uint4 each
    *(uint4*)&At[r][q * 8] =
        *(const uint4*)(A + (size_t)(m0 + r) * ldK + kc * 32 + q * 8);
}
// B tile (transposed): 128x32 bf16 from BwT[n][K], cols kc*32..
__device__ __forceinline__ void stage_Bt(bf16_t (*BtT)[32], const bf16_t* BwT, int ldK,
                                         int kc, int tid) {
#pragma unroll
    for (int t = 0; t < 2; ++t) {                        // 512 uint4 total
        const int i = tid + t * 256;
        const int n = i >> 2, q = i & 3;
        *(uint4*)&BtT[n][q * 8] =
            *(const uint4*)(BwT + (size_t)n * ldK + kc * 32 + q * 8);
    }
}

// ---------------------------------------------------------------------------
// Tiled bf16 WMMA GEMM: Out[4096,128] = A[4096,K] @ B[K,128] (+ bias)
// BwT passed transposed [128][K]. One WG = 64x128 output tile; 8 waves,
// wave w owns N-tile column w.
// ---------------------------------------------------------------------------
__global__ __launch_bounds__(256) void gemm_bias_kernel(const bf16_t* __restrict__ A,
                                                        const bf16_t* __restrict__ BwT,
                                                        const float* __restrict__ bias,
                                                        bf16_t* __restrict__ Out, int K) {
    __shared__ alignas(16) bf16_t At[64][32];
    __shared__ alignas(16) bf16_t BtT[128][32];
    const int tid = threadIdx.x, lane = tid & 31, wave = tid >> 5;
    const int m0 = blockIdx.x * 64;
    const v8f zero = {0.f, 0.f, 0.f, 0.f, 0.f, 0.f, 0.f, 0.f};
    v8f acc[4];
#pragma unroll
    for (int i = 0; i < 4; ++i) acc[i] = zero;

    const int nchunk = K >> 5;
    for (int kc = 0; kc < nchunk; ++kc) {
        stage_A(At, A, K, m0, kc, tid);
        stage_Bt(BtT, BwT, K, kc, tid);
        if (kc + 1 < nchunk) __builtin_prefetch(BwT + (kc + 1) * 32, 0, 1);
        __syncthreads();
        const v16bf bfg = load_bfrag_t(&BtT[0][0], wave * 16, lane);
#pragma unroll
        for (int mt = 0; mt < 4; ++mt) {
            const v16bf af = load_afrag(&At[0][0], 32, mt * 16, lane);
            acc[mt] = wmma_bf16(af, bfg, acc[mt]);
        }
        __syncthreads();
    }
    const int n = wave * 16 + (lane & 15), hh = lane >> 4;
    const float bv = bias ? bias[n] : 0.f;
#pragma unroll
    for (int mt = 0; mt < 4; ++mt)
#pragma unroll
        for (int v = 0; v < 8; ++v) {
            const int m = m0 + mt * 16 + v + 8 * hh;
            Out[(size_t)m * Ff + n] = f2b(acc[mt][v] + bv);
        }
}

// ---------------------------------------------------------------------------
// Fused CFConv: one WG per atom (grid = B*A). GEMM1 (64x32 @ 32x128) -> ssp ->
// GEMM2 (64x128 @ 128x128) -> +b2, * (cutoff*mask) * gathered y_j -> reduce
// over 64 neighbors via LDS float atomics -> agg[128].
// ---------------------------------------------------------------------------
__global__ __launch_bounds__(256) void cfconv_kernel(const bf16_t* __restrict__ fij,
                                                     const bf16_t* __restrict__ w1T,
                                                     const float* __restrict__ b1,
                                                     const bf16_t* __restrict__ w2T,
                                                     const float* __restrict__ b2,
                                                     const bf16_t* __restrict__ y,
                                                     const int* __restrict__ nbr,
                                                     const float* __restrict__ scal,
                                                     bf16_t* __restrict__ aggOut) {
    __shared__ alignas(16) bf16_t Af[64][32];     // gaussian features (A of GEMM1)
    __shared__ alignas(16) bf16_t Ht[64][128];    // ssp(GEMM1) hidden (A of GEMM2)
    __shared__ alignas(16) bf16_t Yj[64][128];    // gathered neighbor features
    __shared__ alignas(16) bf16_t BtT[128][32];   // staged transposed weight chunk
    __shared__ float  aggL[128];
    __shared__ float  scl[64];

    const int tid = threadIdx.x, lane = tid & 31, wave = tid >> 5;
    const int row = blockIdx.x;              // b*A + a
    const int b   = row >> 9;
    const size_t p0 = (size_t)row * Nn;

    // fij tile: 2048 bf16 contiguous -> 256 uint4, 1 per thread
    *(uint4*)(&Af[0][0] + tid * 8) = *(const uint4*)(fij + p0 * GPAD + tid * 8);
    if (tid < 64)  scl[tid]  = scal[p0 + tid];
    if (tid < 128) aggL[tid] = 0.f;
    {   // gather y_j: thread -> (row tid>>2, 32-col quarter tid&3), 4 uint4 each
        const int r = tid >> 2, q = tid & 3;
        const int j = nbr[p0 + r];
        const bf16_t* src = y + (size_t)(b * An + j) * Ff + q * 32;
#pragma unroll
        for (int c = 0; c < 4; ++c)
            *(uint4*)&Yj[r][q * 32 + c * 8] = *(const uint4*)(src + c * 8);
    }
    stage_Bt(BtT, w1T, GPAD, 0, tid);        // w1T is [128][32]
    __syncthreads();

    const int n = wave * 16 + (lane & 15), hh = lane >> 4;

    // ---- GEMM1 + ssp epilogue into Ht -------------------------------------
    {
        const v16bf bfg = load_bfrag_t(&BtT[0][0], wave * 16, lane);
        const float b1v = b1[n];
#pragma unroll
        for (int mt = 0; mt < 4; ++mt) {
            const v16bf af = load_afrag(&Af[0][0], 32, mt * 16, lane);
            v8f acc = {0.f, 0.f, 0.f, 0.f, 0.f, 0.f, 0.f, 0.f};
            acc = wmma_bf16(af, bfg, acc);
#pragma unroll
            for (int v = 0; v < 8; ++v) {
                const int m = mt * 16 + v + 8 * hh;
                Ht[m][n] = f2b(ssp(acc[v] + b1v));
            }
        }
    }
    __syncthreads();

    // ---- GEMM2 (K=128 in 4 chunks; A lives in LDS already) ----------------
    const v8f zero = {0.f, 0.f, 0.f, 0.f, 0.f, 0.f, 0.f, 0.f};
    v8f acc2[4];
#pragma unroll
    for (int i = 0; i < 4; ++i) acc2[i] = zero;
#pragma unroll
    for (int kc = 0; kc < 4; ++kc) {
        stage_Bt(BtT, w2T, Ff, kc, tid);     // w2T is [128][128]
        if (kc + 1 < 4) __builtin_prefetch(w2T + (kc + 1) * 32, 0, 1);
        __syncthreads();
        const v16bf bfg = load_bfrag_t(&BtT[0][0], wave * 16, lane);
#pragma unroll
        for (int mt = 0; mt < 4; ++mt) {
            const v16bf af = load_afrag(&Ht[0][0] + kc * 32, 128, mt * 16, lane);
            acc2[mt] = wmma_bf16(af, bfg, acc2[mt]);
        }
        __syncthreads();
    }

    // ---- epilogue: W = acc2 + b2; agg[n] = sum_m W*scale[m]*y_j[m][n] -----
    const float b2v = b2[n];
#pragma unroll
    for (int mt = 0; mt < 4; ++mt)
#pragma unroll
        for (int v = 0; v < 8; ++v) {
            const int m = mt * 16 + v + 8 * hh;
            const float Wv = acc2[mt][v] + b2v;
            atomicAdd(&aggL[n], Wv * scl[m] * b2f(Yj[m][n]));
        }
    __syncthreads();
    if (tid < 128) aggOut[(size_t)row * Ff + tid] = f2b(aggL[tid]);
}

// ---------------------------------------------------------------------------
// Update: acc = t@dense (K=128) + G@ang (K=512); x += ssp(acc + dense_b).
// DwT [128][128], AwT [128][512] (transposed).
// ---------------------------------------------------------------------------
__global__ __launch_bounds__(256) void update_kernel(const bf16_t* __restrict__ T,
                                                     const bf16_t* __restrict__ DwT,
                                                     const float* __restrict__ db,
                                                     const bf16_t* __restrict__ Gm,
                                                     const bf16_t* __restrict__ AwT,
                                                     float* __restrict__ xf,
                                                     bf16_t* __restrict__ xb) {
    __shared__ alignas(16) bf16_t At[64][32];
    __shared__ alignas(16) bf16_t BtT[128][32];
    const int tid = threadIdx.x, lane = tid & 31, wave = tid >> 5;
    const int m0 = blockIdx.x * 64;
    const v8f zero = {0.f, 0.f, 0.f, 0.f, 0.f, 0.f, 0.f, 0.f};
    v8f acc[4];
#pragma unroll
    for (int i = 0; i < 4; ++i) acc[i] = zero;

    // pass 1: T [*,128] @ dense
#pragma unroll
    for (int kc = 0; kc < 4; ++kc) {
        stage_A(At, T, 128, m0, kc, tid);
        stage_Bt(BtT, DwT, Ff, kc, tid);
        __syncthreads();
        const v16bf bfg = load_bfrag_t(&BtT[0][0], wave * 16, lane);
#pragma unroll
        for (int mt = 0; mt < 4; ++mt) {
            const v16bf af = load_afrag(&At[0][0], 32, mt * 16, lane);
            acc[mt] = wmma_bf16(af, bfg, acc[mt]);
        }
        __syncthreads();
    }
    // pass 2: Gm [*,512] @ ang into the same accumulators
    for (int kc = 0; kc < 16; ++kc) {
        stage_A(At, Gm, GAn, m0, kc, tid);
        stage_Bt(BtT, AwT, GAn, kc, tid);
        if (kc + 1 < 16) __builtin_prefetch(AwT + (kc + 1) * 32, 0, 1);
        __syncthreads();
        const v16bf bfg = load_bfrag_t(&BtT[0][0], wave * 16, lane);
#pragma unroll
        for (int mt = 0; mt < 4; ++mt) {
            const v16bf af = load_afrag(&At[0][0], 32, mt * 16, lane);
            acc[mt] = wmma_bf16(af, bfg, acc[mt]);
        }
        __syncthreads();
    }
    const int n = wave * 16 + (lane & 15), hh = lane >> 4;
    const float dbv = db[n];
#pragma unroll
    for (int mt = 0; mt < 4; ++mt)
#pragma unroll
        for (int v = 0; v < 8; ++v) {
            const int m = m0 + mt * 16 + v + 8 * hh;
            const float xv = xf[(size_t)m * Dd + n] + ssp(acc[mt][v] + dbv);
            xf[(size_t)m * Dd + n] = xv;
            xb[(size_t)m * Dd + n] = f2b(xv);
        }
}

// ---------------------------------------------------------------------------
// Host launcher
// ---------------------------------------------------------------------------
extern "C" void kernel_launch(void* const* d_in, const int* in_sizes, int n_in,
                              void* d_out, int out_size, void* d_ws, size_t ws_size,
                              hipStream_t stream) {
    (void)in_sizes; (void)n_in; (void)out_size; (void)ws_size;
    const int*   atomic_numbers = (const int*)d_in[0];
    const float* positions      = (const float*)d_in[1];
    const int*   neighbors      = (const int*)d_in[2];
    const int*   neighbor_mask  = (const int*)d_in[3];
    const float* G_i            = (const float*)d_in[4];
    const float* embedding      = (const float*)d_in[5];
    const float* filt_w1        = (const float*)d_in[6];
    const float* filt_b1        = (const float*)d_in[7];
    const float* filt_w2        = (const float*)d_in[8];
    const float* filt_b2        = (const float*)d_in[9];
    const float* in2f_w         = (const float*)d_in[10];
    const float* f2out_w        = (const float*)d_in[11];
    const float* f2out_b        = (const float*)d_in[12];
    const float* dense_w        = (const float*)d_in[13];
    const float* dense_b        = (const float*)d_in[14];
    const float* ang_w          = (const float*)d_in[15];

    float* xf = (float*)d_out;               // x lives in d_out (fp32 master copy)

    char* ws = (char*)d_ws;
    size_t off = 0;
    auto take = [&](size_t bytes) -> void* {
        void* p = ws + off;
        off += (bytes + 255) & ~(size_t)255;
        return p;
    };
    bf16_t* x_bf  = (bf16_t*)take((size_t)BA * Dd * 2);
    bf16_t* fij   = (bf16_t*)take((size_t)BA * Nn * GPAD * 2);   // ~16 MB
    float*  scal  = (float*) take((size_t)BA * Nn * 4);
    bf16_t* y_bf  = (bf16_t*)take((size_t)BA * Ff * 2);
    bf16_t* agg   = (bf16_t*)take((size_t)BA * Ff * 2);
    bf16_t* t_bf  = (bf16_t*)take((size_t)BA * Ff * 2);
    bf16_t* G_bf  = (bf16_t*)take((size_t)BA * GAn * 2);
    bf16_t* w1t   = (bf16_t*)take((size_t)3 * Ff * GPAD * 2);    // [L][128][32]
    bf16_t* w2t   = (bf16_t*)take((size_t)3 * Ff * Ff * 2);      // [L][128][128]
    bf16_t* in2ft = (bf16_t*)take((size_t)3 * Ff * Dd * 2);
    bf16_t* f2ot  = (bf16_t*)take((size_t)3 * Dd * Ff * 2);
    bf16_t* dent  = (bf16_t*)take((size_t)3 * Dd * Dd * 2);
    bf16_t* angt  = (bf16_t*)take((size_t)3 * Dd * GAn * 2);     // [L][128][512]

    // prep
    embed_kernel<<<(BA * Dd) / 256, 256, 0, stream>>>(atomic_numbers, embedding, xf, x_bf);
    dist_gauss_kernel<<<(BA * Nn) / 256, 256, 0, stream>>>(positions, neighbors,
                                                           neighbor_mask, fij, scal);
    cvt_pad_w1t_kernel<<<(3 * Ff * GPAD) / 256, 256, 0, stream>>>(filt_w1, w1t);
    cvt_t_kernel<<<(3 * Ff * Ff + 255) / 256, 256, 0, stream>>>(filt_w2, w2t, Ff, Ff,
                                                                3 * Ff * Ff);
    cvt_t_kernel<<<(3 * Dd * Ff + 255) / 256, 256, 0, stream>>>(in2f_w, in2ft, Dd, Ff,
                                                                3 * Dd * Ff);
    cvt_t_kernel<<<(3 * Ff * Dd + 255) / 256, 256, 0, stream>>>(f2out_w, f2ot, Ff, Dd,
                                                                3 * Ff * Dd);
    cvt_t_kernel<<<(3 * Dd * Dd + 255) / 256, 256, 0, stream>>>(dense_w, dent, Dd, Dd,
                                                                3 * Dd * Dd);
    cvt_t_kernel<<<(3 * GAn * Dd + 255) / 256, 256, 0, stream>>>(ang_w, angt, GAn, Dd,
                                                                 3 * GAn * Dd);
    cvt_kernel<<<(BA * GAn + 255) / 256, 256, 0, stream>>>(G_i, G_bf, BA * GAn);

    for (int l = 0; l < 3; ++l) {
        gemm_bias_kernel<<<BA / 64, 256, 0, stream>>>(x_bf, in2ft + (size_t)l * Ff * Dd,
                                                      nullptr, y_bf, Dd);
        cfconv_kernel<<<BA, 256, 0, stream>>>(fij, w1t + (size_t)l * Ff * GPAD,
                                              filt_b1 + (size_t)l * Ff,
                                              w2t + (size_t)l * Ff * Ff,
                                              filt_b2 + (size_t)l * Ff,
                                              y_bf, neighbors, scal, agg);
        gemm_bias_kernel<<<BA / 64, 256, 0, stream>>>(agg, f2ot + (size_t)l * Dd * Ff,
                                                      f2out_b + (size_t)l * Dd, t_bf, Ff);
        update_kernel<<<BA / 64, 256, 0, stream>>>(t_bf, dent + (size_t)l * Dd * Dd,
                                                   dense_b + (size_t)l * Dd,
                                                   G_bf, angt + (size_t)l * Dd * GAn,
                                                   xf, x_bf);
    }
}
